// FocusedLinearChannelAttention_1821066133520
// MI455X (gfx1250) — compile-verified, matrix operand order checked by reading, not run on hardware
//
#include <hip/hip_runtime.h>
#include <math.h>
#include <stdint.h>

// Problem constants (match reference)
#define B_     4
#define S_     8
#define C_     64
#define HEADS_ 4
#define HD_    16
#define N_     32768
#define BH_    16
#define EPS_   1e-6f

typedef __bf16 bf16_t;
typedef bf16_t   v16bf __attribute__((ext_vector_type(16)));
typedef float    v8f   __attribute__((ext_vector_type(8)));
typedef unsigned v4u   __attribute__((ext_vector_type(4)));
typedef int      v8i   __attribute__((ext_vector_type(8)));
typedef int      v4i   __attribute__((ext_vector_type(4)));

union FragA { v16bf v; unsigned u[8]; };

__device__ __forceinline__ unsigned f2bf(float f) {
  unsigned u = __float_as_uint(f);
  return (u + 0x7FFFu + ((u >> 16) & 1u)) >> 16;   // round-to-nearest-even bf16
}
__device__ __forceinline__ unsigned pack_bf(float lo, float hi) {
  return f2bf(lo) | (f2bf(hi) << 16);
}
// 16-bit A-matrix 16x32 layout (ISA 7.12.2): lane L holds row m=L&15;
// VGPR j<4: K = kh+2j, kh = (L>>4)*8 ; VGPR j>=4: K = 16+kh+2(j-4). Pairs packed.
__device__ __forceinline__ int kpair(int j, int lane) {
  int kh = (lane >> 4) << 3;
  return (j < 4) ? (kh + 2 * j) : (16 + kh + 2 * (j - 4));
}
__device__ __forceinline__ v8f wmma_bf16(const FragA& a, const FragA& b, v8f c) {
  return __builtin_amdgcn_wmma_f32_16x16x32_bf16(false, a.v, false, b.v,
                                                 (short)0, c, false, false);
}

// ------------------------- Tensor Data Mover (TDM) 2D tile load -----------
#if defined(__has_builtin)
#if __has_builtin(__builtin_amdgcn_tensor_load_to_lds) && \
    __has_builtin(__builtin_amdgcn_s_wait_tensorcnt)
#define TDM_OK 1
#endif
#endif

#ifdef TDM_OK
// Generic LDS pointer: low 32 bits are the LDS byte offset (flat aperture).
__device__ __forceinline__ unsigned lds_offset(const void* p) {
  return (unsigned)(uintptr_t)p;
}
// 2D tile load: tile_d0 elements (f32) per row, tile_d1 rows, row stride
// stride_d0 elements; LDS rows padded by (pad_amt_enc+1) DWORDs every
// 2<<pad_int_enc DWORDs (ISA 8.4 encodings).
__device__ __forceinline__ void tdm_load_2d(unsigned lds_off, const void* gaddr,
                                            unsigned tensor_d0, unsigned tensor_d1,
                                            unsigned tile_d0, unsigned tile_d1,
                                            unsigned stride_d0,
                                            unsigned pad_int_enc, unsigned pad_amt_enc) {
  unsigned long long ga = (unsigned long long)(uintptr_t)gaddr;
  v4u g0;
  g0[0] = 1u;                                       // count=1, user descriptor
  g0[1] = lds_off;                                  // lds_addr (bytes)
  g0[2] = (unsigned)ga;                             // global_addr[31:0]
  g0[3] = (unsigned)(ga >> 32) | (2u << 30);        // global_addr[56:32] | type=2
  v8i g1;
  g1[0] = (int)((2u << 16) | (1u << 20) |           // data_size=4B, pad_enable
                (pad_int_enc << 22) | (pad_amt_enc << 25));
  g1[1] = (int)((tensor_d0 & 0xFFFFu) << 16);       // tensor_dim0[15:0] -> bits[63:48]
  g1[2] = (int)((tensor_d0 >> 16) | ((tensor_d1 & 0xFFFFu) << 16));
  g1[3] = (int)((tensor_d1 >> 16) | (tile_d0 << 16));   // tile_dim0 -> bits[127:112]
  g1[4] = (int)(tile_d1 & 0xFFFFu);                 // tile_dim1; tile_dim2=0
  g1[5] = (int)stride_d0;                           // tensor_dim0_stride[31:0]
  g1[6] = 0;                                        // stride hi / dim1_stride
  g1[7] = 0;
  v4i g2 = {0, 0, 0, 0};                            // 2D: higher dims unused
  v4i g3 = {0, 0, 0, 0};
#if __clang_major__ >= 23
  v8i g4 = {0, 0, 0, 0, 0, 0, 0, 0};
  __builtin_amdgcn_tensor_load_to_lds(g0, g1, g2, g3, g4, 0);
#else
  __builtin_amdgcn_tensor_load_to_lds(g0, g1, g2, g3, 0);
#endif
}
#endif  // TDM_OK

// ---------------------------------------------------------------- prep ------
__global__ __launch_bounds__(256) void prep_kernel(
    const float* __restrict__ Wq, const float* __restrict__ Wk,
    const float* __restrict__ Wv, const float* __restrict__ Wp,
    const float* __restrict__ scale,
    unsigned* __restrict__ wtq, unsigned* __restrict__ wtk,
    unsigned* __restrict__ wtv, unsigned* __restrict__ wtp,
    float* __restrict__ sc, float* __restrict__ kv, float* __restrict__ ksum) {
  const int t = threadIdx.x;
  // Transposed bf16 weights: wt[c_out][c_in/2] = pack(W[ci][co], W[ci+1][co])
  for (int idx = t; idx < 2048; idx += 256) {
    int co = idx >> 5;
    int ci = (idx & 31) << 1;
    wtq[idx] = pack_bf(Wq[ci * 64 + co], Wq[(ci + 1) * 64 + co]);
    wtk[idx] = pack_bf(Wk[ci * 64 + co], Wk[(ci + 1) * 64 + co]);
    wtv[idx] = pack_bf(Wv[ci * 64 + co], Wv[(ci + 1) * 64 + co]);
    wtp[idx] = pack_bf(Wp[ci * 64 + co], Wp[(ci + 1) * 64 + co]);
  }
  if (t < 64) {
    float x = scale[t];
    sc[t] = (x > 20.f) ? x : log1pf(expf(x));   // softplus
  }
  for (int idx = t; idx < BH_ * 256; idx += 256) kv[idx] = 0.f;
  if (t < BH_ * HD_) ksum[t] = 0.f;
}

// ----------------------------------------------------- QKV + focusing ------
__global__ __launch_bounds__(128) void qkv_focus_kernel(
    const float* __restrict__ x, const float* __restrict__ pos,
    const unsigned* __restrict__ wtq, const unsigned* __restrict__ wtk,
    const unsigned* __restrict__ wtv, const float* __restrict__ sc_g,
    float* __restrict__ q_ws, float* __restrict__ k_ws, float* __restrict__ v_ws) {
  __shared__ union {
    struct { float x[64 * 67]; unsigned w[3 * 2048]; } in;  // x tile + 3 bf16 weights
    struct { float q[64 * 65]; float k[64 * 65]; } st;      // focusing stage
  } sm;
  __shared__ float sc_l[64];

  const int t  = threadIdx.x;
  const int b  = blockIdx.y;
  const int n0 = blockIdx.x * 64;

#ifdef TDM_OK
  if (t == 0) {   // one TDM descriptor: 64 rows (c) x 64 f32 (n), stride N, pad->67
    tdm_load_2d(lds_offset(&sm.in.x[0]), x + (size_t)b * 64 * N_ + n0,
                /*tensor_d0=*/N_, /*tensor_d1=*/64,
                /*tile_d0=*/64, /*tile_d1=*/64,
                /*stride_d0=*/N_, /*pad 64dw*/5, /*+3dw*/2);
  }
#else
  for (int i = t; i < 64 * 64; i += 128) {       // x[b][c][n0:n0+64], coalesced in n
    int c = i >> 6, col = i & 63;
    sm.in.x[c * 67 + col] = x[((size_t)b * 64 + c) * N_ + n0 + col];
  }
#endif
  for (int i = t; i < 2048; i += 128) {          // overlaps with TDM DMA
    sm.in.w[i]        = wtq[i];
    sm.in.w[2048 + i] = wtk[i];
    sm.in.w[4096 + i] = wtv[i];
  }
  if (t < 64) sc_l[t] = sc_g[t];
#ifdef TDM_OK
  if (t == 0) __builtin_amdgcn_s_wait_tensorcnt(0);
#endif
  __syncthreads();

  const int lane  = t & 31;
  const int wv_id = t >> 5;
  const int tok0  = wv_id * 16;                  // each wave: 16 tokens
  const int m     = lane & 15;

  FragA a0, a1;                                  // A fragments for K=0..31, 32..63
#pragma unroll
  for (int j = 0; j < 8; ++j) {
    int kk = kpair(j, lane);
    a0.u[j] = pack_bf(sm.in.x[kk * 67 + tok0 + m],        sm.in.x[(kk + 1) * 67 + tok0 + m]);
    a1.u[j] = pack_bf(sm.in.x[(32 + kk) * 67 + tok0 + m], sm.in.x[(33 + kk) * 67 + tok0 + m]);
  }

  const v8f zero = {0.f, 0.f, 0.f, 0.f, 0.f, 0.f, 0.f, 0.f};
  v8f aq[4], ak[4], av[4];
#pragma unroll
  for (int nt = 0; nt < 4; ++nt) { aq[nt] = zero; ak[nt] = zero; av[nt] = zero; }

  const int ncol = lane & 15;
#pragma unroll
  for (int nt = 0; nt < 4; ++nt) {
    const int nb = nt * 16;
#pragma unroll
    for (int ks = 0; ks < 2; ++ks) {
      FragA bq, bk, bv;
#pragma unroll
      for (int j = 0; j < 8; ++j) {
        int kk  = ks * 32 + kpair(j, lane);
        int off = (nb + ncol) * 32 + (kk >> 1);  // transposed bf16-pair table
        bq.u[j] = sm.in.w[off];
        bk.u[j] = sm.in.w[2048 + off];
        bv.u[j] = sm.in.w[4096 + off];
      }
      const FragA& a = ks ? a1 : a0;
      aq[nt] = wmma_bf16(a, bq, aq[nt]);
      ak[nt] = wmma_bf16(a, bk, ak[nt]);
      av[nt] = wmma_bf16(a, bv, av[nt]);
    }
  }
  __syncthreads();                                // done reading sm.in

  // Stage q, k(+pos) to LDS; v goes straight out (head-split, coalesced).
  const int half = lane >> 4, col = lane & 15;
#pragma unroll
  for (int nt = 0; nt < 4; ++nt) {
    const int cch = nt * 16 + col;
#pragma unroll
    for (int r = 0; r < 8; ++r) {
      int tok = tok0 + r + 8 * half;
      int n   = n0 + tok;
      sm.st.q[tok * 65 + cch] = aq[nt][r];
      sm.st.k[tok * 65 + cch] = ak[nt][r] + pos[(size_t)n * 64 + cch];
      v_ws[(((size_t)(b * HEADS_ + nt)) * N_ + n) * HD_ + col] = av[nt][r];
    }
  }
  __syncthreads();

  if (t < 64) {                                  // one token per thread
    const int tok = t;
    const int n   = n0 + tok;
    float* qr = &sm.st.q[tok * 65];
    float* kr = &sm.st.k[tok * 65];
    float qn2 = 0.f, kn2 = 0.f;
    for (int c = 0; c < 64; ++c) {
      float qv = (fmaxf(qr[c], 0.f) + EPS_) / sc_l[c];
      float kv = (fmaxf(kr[c], 0.f) + EPS_) / sc_l[c];
      qr[c] = qv; kr[c] = kv;
      qn2 += qv * qv; kn2 += kv * kv;
    }
    float qn6 = 0.f, kn6 = 0.f;
    for (int c = 0; c < 64; ++c) {
      float q3 = qr[c] * qr[c] * qr[c];
      float k3 = kr[c] * kr[c] * kr[c];
      qn6 += q3 * q3; kn6 += k3 * k3;
    }
    float qf = sqrtf(qn2) * rsqrtf(qn6);          // |q| / |q^3|
    float kf = sqrtf(kn2) * rsqrtf(kn6);
    for (int h = 0; h < 4; ++h) {
      size_t base = (((size_t)(b * HEADS_ + h)) * N_ + n) * HD_;
      for (int d = 0; d < 16; ++d) {
        int c = h * 16 + d;
        float qv = qr[c], kv = kr[c];
        q_ws[base + d] = qv * qv * qv * qf;
        k_ws[base + d] = kv * kv * kv * kf;
      }
    }
  }
}

// ------------------------------------------- kv = k^T v, ksum (WMMA) ------
__global__ __launch_bounds__(256) void kv_kernel(
    const float* __restrict__ k_ws, const float* __restrict__ v_ws,
    float* __restrict__ kv_g, float* __restrict__ ksum_g) {
  __shared__ float kbuf[8][32][17];
  __shared__ float vbuf[8][32][17];
  const int t    = threadIdx.x;
  const int lane = t & 31;
  const int w    = t >> 5;
  const int bh   = blockIdx.y;
  const int base = blockIdx.x * 4096 + w * 512;

  FragA bones;                                    // all-ones B -> column sums of k
#pragma unroll
  for (int j = 0; j < 8; ++j) bones.u[j] = 0x3F803F80u;

  v8f acc  = {0.f, 0.f, 0.f, 0.f, 0.f, 0.f, 0.f, 0.f};
  v8f acc1 = acc;
  const int cc = lane & 15;
  for (int it = 0; it < 16; ++it) {
    const int n0 = base + it * 32;
    const float* kp = k_ws + ((size_t)bh * N_ + n0 + lane) * HD_;
    const float* vp = v_ws + ((size_t)bh * N_ + n0 + lane) * HD_;
#pragma unroll
    for (int c = 0; c < 16; ++c) { kbuf[w][lane][c] = kp[c]; vbuf[w][lane][c] = vp[c]; }
    FragA a, bv;                                  // A = k^T [16c x 32tok], B = v [32tok x 16d]
#pragma unroll
    for (int j = 0; j < 8; ++j) {
      int kk  = kpair(j, lane);
      a.u[j]  = pack_bf(kbuf[w][kk][cc], kbuf[w][kk + 1][cc]);
      bv.u[j] = pack_bf(vbuf[w][kk][cc], vbuf[w][kk + 1][cc]);
    }
    acc  = wmma_bf16(a, bv, acc);
    acc1 = wmma_bf16(a, bones, acc1);
  }
  const int half = lane >> 4, col = lane & 15;
#pragma unroll
  for (int r = 0; r < 8; ++r)
    atomicAdd(&kv_g[bh * 256 + (r + 8 * half) * 16 + col], acc[r]);
  if (col == 0) {
#pragma unroll
    for (int r = 0; r < 8; ++r)
      atomicAdd(&ksum_g[bh * 16 + r + 8 * half], acc1[r]);
  }
}

// ----------------------------------------- out = (q@kv)*z -> mid ----------
__global__ __launch_bounds__(256) void attn_kernel(
    const float* __restrict__ q_ws, const float* __restrict__ kv_g,
    const float* __restrict__ ksum_g, float* __restrict__ mid) {
  __shared__ float kvl[4][16][16];
  __shared__ float ksl[4][16];
  const int t = threadIdx.x;
  const int b = blockIdx.y;
  const int n = blockIdx.x * 256 + t;
  for (int i = t; i < 4 * 256; i += 256) {
    int h = i >> 8, rest = i & 255;
    kvl[h][rest >> 4][rest & 15] = kv_g[(b * 4 + h) * 256 + rest];
  }
  if (t < 64) ksl[t >> 4][t & 15] = ksum_g[b * 64 + t];
  __syncthreads();

  for (int h = 0; h < 4; ++h) {
    const float* qp = q_ws + (((size_t)(b * 4 + h)) * N_ + n) * HD_;
    float qv[16];
#pragma unroll
    for (int c = 0; c < 16; ++c) qv[c] = qp[c];
    float dot = 0.f;
#pragma unroll
    for (int c = 0; c < 16; ++c) dot += qv[c] * ksl[h][c];
    float z = 1.f / (dot + EPS_);
    float* op = mid + ((size_t)b * N_ + n) * 64 + h * 16;
#pragma unroll
    for (int d = 0; d < 16; ++d) {
      float o = 0.f;
#pragma unroll
      for (int c = 0; c < 16; ++c) o += qv[c] * kvl[h][c][d];
      op[d] = o * z;
    }
  }
}

// ----------------------------------- depthwise 5x5x5 conv, += into mid ----
__global__ __launch_bounds__(256) void dwconv_kernel(
    const float* __restrict__ v_ws, const float* __restrict__ dwc_w,
    const float* __restrict__ dwc_b, float* __restrict__ mid) {
  __shared__ float vh[8][12][12][8];   // full S, 8+4 halo in a,b, 8 d-channels
  __shared__ float wl[8][125];
  __shared__ float bl[8];
  const int t  = threadIdx.x;
  const int bh = blockIdx.z;
  const int b  = bh >> 2, h = bh & 3;
  const int a0 = blockIdx.y * 8;
  const int b0 = blockIdx.x * 8;

  for (int dh = 0; dh < 2; ++dh) {
    for (int i = t; i < 8 * 12 * 12 * 8; i += 256) {
      int dsub = i & 7;
      int rest = i >> 3;
      int br = rest % 12, ar = (rest / 12) % 12, s = rest / 144;
      int aa = a0 + ar - 2, bb = b0 + br - 2;
      float val = 0.f;
      if (aa >= 0 && aa < 64 && bb >= 0 && bb < 64) {
        int n = s * 4096 + aa * 64 + bb;
        val = v_ws[((size_t)bh * N_ + n) * HD_ + dh * 8 + dsub];
      }
      vh[s][ar][br][dsub] = val;
    }
    for (int i = t; i < 8 * 125; i += 256)
      wl[i / 125][i % 125] = dwc_w[(dh * 8 + i / 125) * 125 + i % 125];
    if (t < 8) bl[t] = dwc_b[dh * 8 + t];
    __syncthreads();

    for (int i = t; i < 4096; i += 256) {        // 8s x 8a x 8b x 8d outputs
      int dsub = i & 7;
      int bb = (i >> 3) & 7;
      int aa = (i >> 6) & 7;
      int s  = (i >> 9);
      float sum = bl[dsub];
      for (int kd = 0; kd < 5; ++kd) {
        int ss = s + kd - 2;
        if (ss < 0 || ss >= 8) continue;         // SAME zero padding in s
#pragma unroll
        for (int kh = 0; kh < 5; ++kh)
#pragma unroll
          for (int kw = 0; kw < 5; ++kw)
            sum += wl[dsub][kd * 25 + kh * 5 + kw] * vh[ss][aa + kh][bb + kw][dsub];
      }
      int n = s * 4096 + (a0 + aa) * 64 + (b0 + bb);
      mid[((size_t)b * N_ + n) * 64 + h * 16 + dh * 8 + dsub] += sum;
    }
    __syncthreads();
  }
}

// --------------------------------- out = mid@Wp + bp, transpose store ----
__global__ __launch_bounds__(128) void proj_kernel(
    const float* __restrict__ mid, const unsigned* __restrict__ wtp,
    const float* __restrict__ bp, float* __restrict__ out) {
  __shared__ union { float x[64 * 65]; float o[64 * 65]; } sm;  // [tok][c], pad->65
  __shared__ unsigned wl[2048];
  __shared__ float bpl[64];
  const int t  = threadIdx.x;
  const int b  = blockIdx.y;
  const int n0 = blockIdx.x * 64;

#ifdef TDM_OK
  if (t == 0) {   // 64 rows (tok) x 64 f32 (c), contiguous rows, pad->65
    tdm_load_2d(lds_offset(&sm.x[0]), mid + ((size_t)b * N_ + n0) * 64,
                /*tensor_d0=*/64, /*tensor_d1=*/N_,
                /*tile_d0=*/64, /*tile_d1=*/64,
                /*stride_d0=*/64, /*pad 64dw*/5, /*+1dw*/0);
  }
#else
  for (int i = t; i < 64 * 64; i += 128) {       // mid[n][c] -> LDS [tok][c]
    int tok = i >> 6, c = i & 63;
    sm.x[tok * 65 + c] = mid[((size_t)b * N_ + n0 + tok) * 64 + c];
  }
#endif
  for (int i = t; i < 2048; i += 128) wl[i] = wtp[i];
  if (t < 64) bpl[t] = bp[t];
#ifdef TDM_OK
  if (t == 0) __builtin_amdgcn_s_wait_tensorcnt(0);
#endif
  __syncthreads();

  const int lane = t & 31, wv_id = t >> 5, tok0 = wv_id * 16, m = lane & 15;
  const int rowbase = (tok0 + m) * 65;           // A row-major in LDS
  FragA a0, a1;
#pragma unroll
  for (int j = 0; j < 8; ++j) {
    int kk = kpair(j, lane);
    a0.u[j] = pack_bf(sm.x[rowbase + kk],      sm.x[rowbase + kk + 1]);
    a1.u[j] = pack_bf(sm.x[rowbase + 32 + kk], sm.x[rowbase + 33 + kk]);
  }
  const v8f zero = {0.f, 0.f, 0.f, 0.f, 0.f, 0.f, 0.f, 0.f};
  v8f acc[4];
  const int ncol = lane & 15;
#pragma unroll
  for (int nt = 0; nt < 4; ++nt) {
    acc[nt] = zero;
#pragma unroll
    for (int ks = 0; ks < 2; ++ks) {
      FragA bw;
#pragma unroll
      for (int j = 0; j < 8; ++j) {
        int kk = ks * 32 + kpair(j, lane);
        bw.u[j] = wl[(nt * 16 + ncol) * 32 + (kk >> 1)];
      }
      acc[nt] = wmma_bf16(ks ? a1 : a0, bw, acc[nt]);
    }
  }
  __syncthreads();                               // done with sm.x

  const int half = lane >> 4, col = lane & 15;
#pragma unroll
  for (int nt = 0; nt < 4; ++nt) {
    int cch = nt * 16 + col;
    float bias = bpl[cch];
#pragma unroll
    for (int r = 0; r < 8; ++r)
      sm.o[(tok0 + r + 8 * half) * 65 + cch] = acc[nt][r] + bias;
  }
  __syncthreads();
  for (int i = t; i < 64 * 64; i += 128) {       // coalesced [b][c][n] store
    int c = i >> 6, col2 = i & 63;
    out[((size_t)b * 64 + c) * N_ + n0 + col2] = sm.o[col2 * 65 + c];
  }
}

// --------------------------------------------------------------- launch ---
extern "C" void kernel_launch(void* const* d_in, const int* in_sizes, int n_in,
                              void* d_out, int out_size, void* d_ws, size_t ws_size,
                              hipStream_t stream) {
  (void)in_sizes; (void)n_in; (void)out_size; (void)ws_size;
  const float* x     = (const float*)d_in[0];
  const float* Wq    = (const float*)d_in[1];
  const float* Wk    = (const float*)d_in[2];
  const float* Wv    = (const float*)d_in[3];
  const float* Wp    = (const float*)d_in[4];
  const float* bp    = (const float*)d_in[5];
  const float* scale = (const float*)d_in[6];
  const float* pos   = (const float*)d_in[7];
  const float* dwcw  = (const float*)d_in[8];
  const float* dwcb  = (const float*)d_in[9];

  char* ws = (char*)d_ws;
  const size_t BIG = (size_t)BH_ * N_ * HD_ * sizeof(float);   // 33.5 MB
  unsigned* wtq  = (unsigned*)(ws + 0);
  unsigned* wtk  = (unsigned*)(ws + 8192);
  unsigned* wtv  = (unsigned*)(ws + 16384);
  unsigned* wtp  = (unsigned*)(ws + 24576);
  float* sc      = (float*)(ws + 32768);
  float* ksum    = (float*)(ws + 33024);
  float* kvb     = (float*)(ws + 34048);
  float* q_ws    = (float*)(ws + 65536);
  float* k_ws    = (float*)(ws + 65536 + BIG);
  float* v_ws    = (float*)(ws + 65536 + 2 * BIG);
  float* mid     = k_ws;   // k dead after kv_kernel -> overlay mid (saves 33.5 MB)

  prep_kernel<<<1, 256, 0, stream>>>(Wq, Wk, Wv, Wp, scale, wtq, wtk, wtv, wtp,
                                     sc, kvb, ksum);
  qkv_focus_kernel<<<dim3(N_ / 64, B_), 128, 0, stream>>>(x, pos, wtq, wtk, wtv,
                                                          sc, q_ws, k_ws, v_ws);
  kv_kernel<<<dim3(8, BH_), 256, 0, stream>>>(k_ws, v_ws, kvb, ksum);
  attn_kernel<<<dim3(N_ / 256, B_), 256, 0, stream>>>(q_ws, kvb, ksum, mid);
  dwconv_kernel<<<dim3(8, 8, BH_), 256, 0, stream>>>(v_ws, dwcw, dwcb, mid);
  proj_kernel<<<dim3(N_ / 64, B_), 128, 0, stream>>>(mid, wtp, bp, (float*)d_out);
}
